// TransformerXL_85272280694996
// MI455X (gfx1250) — compile-verified, matrix operand order checked by reading, not run on hardware
//
#include <hip/hip_runtime.h>
#include <hip/hip_bf16.h>
#include <cstdint>

// ---------------------------------------------------------------------------
// Types
// ---------------------------------------------------------------------------
typedef __bf16 bf16_t;
typedef __attribute__((ext_vector_type(16))) __bf16 v16bf;
typedef __attribute__((ext_vector_type(8)))  __bf16 v8bf;
typedef __attribute__((ext_vector_type(8)))  float  v8f;
typedef __attribute__((ext_vector_type(4)))  int    v4i;

#define BM 128
#define BN 64
#define BK 32
#define ASTR 40   // LDS row stride (bf16 elems): 80B, 16B aligned, conflict-free
#define BSTR 40

// gfx1250 async global->LDS path (ASYNCcnt), with safe fallback.
#if defined(__gfx1250__) && __has_builtin(__builtin_amdgcn_global_load_async_to_lds_b128) && __has_builtin(__builtin_amdgcn_s_wait_asynccnt)
#define USE_ASYNC_LDS 1
#else
#define USE_ASYNC_LDS 0
#endif

#if USE_ASYNC_LDS
typedef __attribute__((address_space(1))) v4i gv4i;   // global int4
typedef __attribute__((address_space(3))) v4i lv4i;   // LDS int4
__device__ __forceinline__ void async_ld16(const void* gsrc, void* ldst) {
  // generic->global is numerically identity; low 32 bits of a generic LDS
  // pointer are the LDS byte offset (flat aperture rules).
  __builtin_amdgcn_global_load_async_to_lds_b128(
      (gv4i*)(unsigned long long)(uintptr_t)gsrc,
      (lv4i*)(unsigned)(uintptr_t)ldst, 0, 0);
}
#endif

__device__ __forceinline__ v16bf cat8(v8bf lo, v8bf hi) {
  return __builtin_shufflevector(lo, hi, 0,1,2,3,4,5,6,7,8,9,10,11,12,13,14,15);
}

// ---------------------------------------------------------------------------
// Generic bf16 WMMA GEMM: C[M,N] = A[M,K] * Bt[N,K]^T   (Bt is N-major)
// Block = 256 threads = 8 waves; block tile 128x64; wave tile 32x32.
// Optional batching via blockIdx.z with per-operand strides (mod for reuse).
// Epilogue policy receives (val, m, n, batch).
// ---------------------------------------------------------------------------
template <class Epi>
__global__ __launch_bounds__(256)
void gemm_k(const bf16_t* __restrict__ A, int lda, long long sAb, int modA,
            const bf16_t* __restrict__ Bt, int ldb, long long sBb, int modB,
            int K, typename Epi::Params ep)
{
  __shared__ __align__(16) bf16_t As[BM * ASTR];
  __shared__ __align__(16) bf16_t Bs[BN * BSTR];

  const int tid = threadIdx.x;
  const int bat = blockIdx.z;
  A  += (long long)(bat % modA) * sAb + (long long)blockIdx.y * BM * lda;
  Bt += (long long)(bat % modB) * sBb + (long long)blockIdx.x * BN * ldb;

  const int wave = tid >> 5;
  const int lane = tid & 31;
  const int hlf  = lane >> 4;      // lane group: selects K half per ISA layout
  const int lrow = lane & 15;
  const int wm   = (wave & 3) << 5;   // 0,32,64,96
  const int wn   = (wave >> 2) << 5;  // 0,32
  const int lr   = tid >> 2;          // 0..63
  const int lc   = (tid & 3) << 3;    // 0,8,16,24
  const int koff = hlf << 3;          // 0 or 8

  v8f acc[2][2] = {};

  for (int k0 = 0; k0 < K; k0 += BK) {
    __syncthreads();
#if USE_ASYNC_LDS
    async_ld16(&A[(long long)lr * lda + k0 + lc],        &As[lr * ASTR + lc]);
    async_ld16(&A[(long long)(lr + 64) * lda + k0 + lc], &As[(lr + 64) * ASTR + lc]);
    async_ld16(&Bt[(long long)lr * ldb + k0 + lc],       &Bs[lr * BSTR + lc]);
    if (k0 + BK < K) {
      __builtin_prefetch(&A[(long long)lr * lda + k0 + BK + lc], 0, 1);
      __builtin_prefetch(&Bt[(long long)lr * ldb + k0 + BK + lc], 0, 1);
    }
    __builtin_amdgcn_s_wait_asynccnt(0);
#else
    *(v8bf*)&As[lr * ASTR + lc]        = *(const v8bf*)&A[(long long)lr * lda + k0 + lc];
    *(v8bf*)&As[(lr + 64) * ASTR + lc] = *(const v8bf*)&A[(long long)(lr + 64) * lda + k0 + lc];
    *(v8bf*)&Bs[lr * BSTR + lc]        = *(const v8bf*)&Bt[(long long)lr * ldb + k0 + lc];
    if (k0 + BK < K) {
      __builtin_prefetch(&A[(long long)lr * lda + k0 + BK + lc], 0, 1);
      __builtin_prefetch(&Bt[(long long)lr * ldb + k0 + BK + lc], 0, 1);
    }
#endif
    __syncthreads();

#pragma unroll
    for (int fi = 0; fi < 2; ++fi) {
      const bf16_t* ap = &As[(wm + fi * 16 + lrow) * ASTR + koff];
      v16bf afr = cat8(*(const v8bf*)ap, *(const v8bf*)(ap + 16));
#pragma unroll
      for (int fj = 0; fj < 2; ++fj) {
        const bf16_t* bp = &Bs[(wn + fj * 16 + lrow) * BSTR + koff];
        v16bf bfr = cat8(*(const v8bf*)bp, *(const v8bf*)(bp + 16));
        acc[fi][fj] = __builtin_amdgcn_wmma_f32_16x16x32_bf16(
            false, afr, false, bfr, (short)0, acc[fi][fj], false, false);
      }
    }
  }

  // C layout: VGPR r -> M = r + 8*(lane>=16), N = lane&15
  const int m0 = blockIdx.y * BM + wm + hlf * 8;
  const int n0 = blockIdx.x * BN + wn + lrow;
#pragma unroll
  for (int fi = 0; fi < 2; ++fi)
#pragma unroll
    for (int fj = 0; fj < 2; ++fj)
#pragma unroll
      for (int r = 0; r < 8; ++r)
        Epi::store(ep, acc[fi][fj][r], m0 + fi * 16 + r, n0 + fj * 16, bat);
}

// ---------------------------------------------------------------------------
// Epilogue policies
// ---------------------------------------------------------------------------
struct EpiF32 {              // f32 store, optional bias, optional batch stride
  struct Params { float* C; int ldc; long long sCb; const float* bias; };
  static __device__ __forceinline__ void store(const Params& p, float v, int m, int n, int bat) {
    if (p.bias) v += p.bias[n];
    p.C[(long long)bat * p.sCb + (long long)m * p.ldc + n] = v;
  }
};

struct EpiBF16 {             // bf16 store
  struct Params { bf16_t* C; int ldc; };
  static __device__ __forceinline__ void store(const Params& p, float v, int m, int n, int bat) {
    (void)bat;
    p.C[(long long)m * p.ldc + n] = (bf16_t)v;
  }
};

struct EpiReluBias {         // relu(v + bias) -> bf16
  struct Params { bf16_t* C; int ldc; const float* bias; };
  static __device__ __forceinline__ void store(const Params& p, float v, int m, int n, int bat) {
    (void)bat;
    v += p.bias[n];
    p.C[(long long)m * p.ldc + n] = (bf16_t)(v > 0.f ? v : 0.f);
  }
};

// QKV scatter. GEMM rows m = j*32 + b (j in [0,512), b in [0,32)), cols n in [0,1536).
// q (n<512, j>=256): qu/qv [b*8+h][i][d] with +u_bias/+v_bias
// k (512<=n<1024):   kk [b*8+h][j][d]
// v (n>=1024):       vT [b*8+h][d][j]  (transposed, ready as B-operand)
struct EpiQKV {
  struct Params { bf16_t *qu, *qv, *kk, *vT; const float *ub, *vb; };
  static __device__ __forceinline__ void store(const Params& p, float v, int m, int n, int bat) {
    (void)bat;
    const int j = m >> 5, b = m & 31;
    if (n < 512) {
      if (j >= 256) {
        const int i = j - 256;
        const long long o = ((long long)(b * 8 + (n >> 6)) * 256 + i) * 64 + (n & 63);
        p.qu[o] = (bf16_t)(v + p.ub[n]);
        p.qv[o] = (bf16_t)(v + p.vb[n]);
      }
    } else if (n < 1024) {
      const int c = n - 512;
      p.kk[((long long)(b * 8 + (c >> 6)) * 512 + j) * 64 + (c & 63)] = (bf16_t)v;
    } else {
      const int c = n - 1024;
      p.vT[((long long)(b * 8 + (c >> 6)) * 64 + (c & 63)) * 512 + j] = (bf16_t)v;
    }
  }
};

// TransformerXL rel-shift accumulate: raw BD (i=m, jj=n) lands at j = jj + i - 255.
// Unique (i,jj)->(i,j) map; score already holds AC (prior kernel), so plain +=.
struct EpiBD {
  struct Params { float* score; };
  static __device__ __forceinline__ void store(const Params& p, float v, int m, int n, int bat) {
    const int j = n + m - 255;
    if (j >= 0 && j < 512)
      p.score[((long long)bat * 256 + m) * 512 + j] += v;
  }
};

// attn@V output (per batch bh = b*8+h): C[i=m, d=n] -> ao rows (i*32+b), col h*64+d
struct EpiAO {
  struct Params { bf16_t* ao; };
  static __device__ __forceinline__ void store(const Params& p, float v, int m, int n, int bat) {
    const int b = bat >> 3, h = bat & 7;
    p.ao[((long long)m * 32 + b) * 512 + h * 64 + n] = (bf16_t)v;
  }
};

// ---------------------------------------------------------------------------
// Elementwise kernels
// ---------------------------------------------------------------------------

// LayerNorm over E=512 with optional residual add, relu, f32/bf16 outputs.
__global__ __launch_bounds__(256)
void ln_k(const float* __restrict__ x, const float* __restrict__ res,
          const float* __restrict__ g, const float* __restrict__ b,
          float eps, int relu, float* __restrict__ of, bf16_t* __restrict__ ob)
{
  const int row = blockIdx.x, t = threadIdx.x;
  __shared__ float red[8];
  const long long base = (long long)row * 512;
  float v0 = x[base + t], v1 = x[base + t + 256];
  if (res) { v0 += res[base + t]; v1 += res[base + t + 256]; }

  float s = v0 + v1;
#pragma unroll
  for (int o = 16; o > 0; o >>= 1) s += __shfl_xor(s, o, 32);
  if ((t & 31) == 0) red[t >> 5] = s;
  __syncthreads();
  float tot = 0.f;
#pragma unroll
  for (int k = 0; k < 8; ++k) tot += red[k];
  const float mean = tot * (1.f / 512.f);
  __syncthreads();

  const float d0 = v0 - mean, d1 = v1 - mean;
  float q = d0 * d0 + d1 * d1;
#pragma unroll
  for (int o = 16; o > 0; o >>= 1) q += __shfl_xor(q, o, 32);
  if ((t & 31) == 0) red[t >> 5] = q;
  __syncthreads();
  float vtot = 0.f;
#pragma unroll
  for (int k = 0; k < 8; ++k) vtot += red[k];
  const float rinv = rsqrtf(vtot * (1.f / 512.f) + eps);

  float o0 = d0 * rinv * g[t] + b[t];
  float o1 = d1 * rinv * g[t + 256] + b[t + 256];
  if (relu) { o0 = fmaxf(o0, 0.f); o1 = fmaxf(o1, 0.f); }
  if (of) { of[base + t] = o0; of[base + t + 256] = o1; }
  if (ob) { ob[base + t] = (bf16_t)o0; ob[base + t + 256] = (bf16_t)o1; }
}

// Masked softmax over klen=512; grid=(i=256, bh=256). Writes attn bf16.
__global__ __launch_bounds__(256)
void softmax_k(const float* __restrict__ score, bf16_t* __restrict__ attn, float scale)
{
  const int i = blockIdx.x, bh = blockIdx.y, t = threadIdx.x;
  __shared__ float red[8];
  const long long base = ((long long)bh * 256 + i) * 512;
  const int lim = i + 256;  // j <= lim unmasked (mask: j > i + MEM)

  float s0 = score[base + t] * scale;                      // t <= 255 <= lim always
  float s1 = (t + 256 <= lim) ? score[base + t + 256] * scale : -1e30f;

  float m = fmaxf(s0, s1);
#pragma unroll
  for (int o = 16; o > 0; o >>= 1) m = fmaxf(m, __shfl_xor(m, o, 32));
  if ((t & 31) == 0) red[t >> 5] = m;
  __syncthreads();
  float mx = red[0];
#pragma unroll
  for (int k = 1; k < 8; ++k) mx = fmaxf(mx, red[k]);
  __syncthreads();

  float e0 = __expf(s0 - mx);
  float e1 = (t + 256 <= lim) ? __expf(s1 - mx) : 0.f;
  float s = e0 + e1;
#pragma unroll
  for (int o = 16; o > 0; o >>= 1) s += __shfl_xor(s, o, 32);
  if ((t & 31) == 0) red[t >> 5] = s;
  __syncthreads();
  float tot = 0.f;
#pragma unroll
  for (int k = 0; k < 8; ++k) tot += red[k];
  const float inv = 1.f / tot;
  attn[base + t]       = (bf16_t)(e0 * inv);
  attn[base + t + 256] = (bf16_t)(e1 * inv);
}

// f32 [R,C] -> bf16 [C,R]
__global__ void transpose_bf_k(const float* __restrict__ in, bf16_t* __restrict__ out,
                               int R, int C)
{
  const long long idx = (long long)blockIdx.x * 256 + threadIdx.x;
  if (idx >= (long long)R * C) return;
  const int r = (int)(idx / C), c = (int)(idx % C);
  out[(long long)c * R + r] = (bf16_t)in[idx];
}

// Stem input: X0 [8192, 576], row m = i*32+b; cols [0,512)=x, [512,576)=act_emb
__global__ void build_stem_k(const float* __restrict__ x, const int* __restrict__ action,
                             const float* __restrict__ act_emb, bf16_t* __restrict__ X0)
{
  const long long idx = (long long)blockIdx.x * 256 + threadIdx.x;  // 8192*576
  const int m = (int)(idx / 576), c = (int)(idx % 576);
  const int i = m >> 5, b = m & 31;
  float v;
  if (c < 512) v = x[((long long)b * 256 + i) * 512 + c];           // x[b, t*8+s, e]
  else         v = act_emb[action[b * 32 + (i >> 3)] * 64 + (c - 512)];
  X0[idx] = (bf16_t)v;
}

// cat [16384, 512], row m = j*32+b; j<256 -> mems[l], else cur rows (j-256)*32+b
__global__ void build_cat_k(const float* __restrict__ mems_l, const float* __restrict__ cur,
                            bf16_t* __restrict__ cat)
{
  const long long idx = (long long)blockIdx.x * 256 + threadIdx.x;  // 16384*512
  const int m = (int)(idx >> 9), e = (int)(idx & 511);
  const int j = m >> 5, b = m & 31;
  const float v = (j < 256) ? mems_l[((long long)j * 32 + b) * 512 + e]
                            : cur[((long long)(j - 256) * 32 + b) * 512 + e];
  cat[idx] = (bf16_t)v;
}

// Sinusoidal pos emb, bf16 [512,512]; thread t: col t=sin, col t+256=cos
__global__ void posemb_k(const int* __restrict__ positions, bf16_t* __restrict__ pe)
{
  const int j = blockIdx.x, t = threadIdx.x;
  const float p = (float)positions[j];
  const float inv = __expf(-((float)t * (1.f / 256.f)) * 9.2103403719761836f); // ln(1e4)
  const float ang = p * inv;
  pe[(long long)j * 512 + t]       = (bf16_t)__sinf(ang);
  pe[(long long)j * 512 + t + 256] = (bf16_t)__cosf(ang);
}

// [L,B,E] rows (i*32+b) -> out [B,L,E]
__global__ void writeout_k(const float* __restrict__ cur, float* __restrict__ out)
{
  const long long idx = (long long)blockIdx.x * 256 + threadIdx.x;  // 8192*512
  const int m = (int)(idx >> 9), e = (int)(idx & 511);
  const int i = m >> 5, b = m & 31;
  out[((long long)b * 256 + i) * 512 + e] = cur[idx];
}

// ---------------------------------------------------------------------------
// Host side
// ---------------------------------------------------------------------------
template <class Epi>
static void run_gemm(hipStream_t s, int M, int N, int K, int batches,
                     const bf16_t* A, int lda, long long sAb, int modA,
                     const bf16_t* Bt, int ldb, long long sBb, int modB,
                     typename Epi::Params p)
{
  dim3 grid(N / BN, M / BM, batches);
  gemm_k<Epi><<<grid, dim3(256), 0, s>>>(A, lda, sAb, modA, Bt, ldb, sBb, modB, K, p);
}

extern "C" void kernel_launch(void* const* d_in, const int* in_sizes, int n_in,
                              void* d_out, int out_size, void* d_ws, size_t ws_size,
                              hipStream_t stream)
{
  (void)in_sizes; (void)n_in; (void)out_size; (void)ws_size;

  const float* x         = (const float*)d_in[0];
  const int*   action    = (const int*)d_in[1];
  const int*   positions = (const int*)d_in[3];
  const float* mems      = (const float*)d_in[4];
  const float* act_emb   = (const float*)d_in[5];
  const float* stem_w1   = (const float*)d_in[6];
  const float* s_ln1_g   = (const float*)d_in[7];
  const float* s_ln1_b   = (const float*)d_in[8];
  const float* stem_w2   = (const float*)d_in[9];
  const float* s_ln2_g   = (const float*)d_in[10];
  const float* s_ln2_b   = (const float*)d_in[11];
  const float* lnf_g     = (const float*)d_in[12];
  const float* lnf_b     = (const float*)d_in[13];
  const float* u_bias    = (const float*)d_in[14];
  const float* v_bias    = (const float*)d_in[15];
  const float* Wqkv      = (const float*)d_in[16];
  const float* Wr        = (const float*)d_in[17];
  const float* Wo        = (const float*)d_in[18];
  const float* ln1_g     = (const float*)d_in[19];
  const float* ln1_b     = (const float*)d_in[20];
  const float* W1        = (const float*)d_in[21];
  const float* b1        = (const float*)d_in[22];
  const float* W2        = (const float*)d_in[23];
  const float* b2        = (const float*)d_in[24];
  const float* ln2_g     = (const float*)d_in[25];
  const float* ln2_b     = (const float*)d_in[26];

  // ---- workspace carve-up ----
  char* p = (char*)d_ws;
  auto alloc = [&](size_t bytes) -> char* {
    char* r = p; p += (bytes + 255) & ~(size_t)255; return r;
  };
  bf16_t* w1T   = (bf16_t*)alloc((size_t)512 * 576 * 2);
  bf16_t* w2T   = (bf16_t*)alloc((size_t)512 * 512 * 2);
  bf16_t* qkvT  = (bf16_t*)alloc((size_t)4 * 1536 * 512 * 2);
  bf16_t* wrT   = (bf16_t*)alloc((size_t)4 * 512 * 512 * 2);
  bf16_t* woT   = (bf16_t*)alloc((size_t)4 * 512 * 512 * 2);
  bf16_t* w1lT  = (bf16_t*)alloc((size_t)4 * 2048 * 512 * 2);
  bf16_t* w2lT  = (bf16_t*)alloc((size_t)4 * 512 * 2048 * 2);
  bf16_t* X0    = (bf16_t*)alloc((size_t)8192 * 576 * 2);
  bf16_t* pe    = (bf16_t*)alloc((size_t)512 * 512 * 2);
  bf16_t* rbf   = (bf16_t*)alloc((size_t)512 * 512 * 2);
  bf16_t* catbf = (bf16_t*)alloc((size_t)16384 * 512 * 2);
  bf16_t* qu    = (bf16_t*)alloc((size_t)256 * 256 * 64 * 2);
  bf16_t* qv    = (bf16_t*)alloc((size_t)256 * 256 * 64 * 2);
  bf16_t* kk    = (bf16_t*)alloc((size_t)256 * 512 * 64 * 2);
  bf16_t* vT    = (bf16_t*)alloc((size_t)256 * 64 * 512 * 2);
  float*  score = (float*)alloc((size_t)256 * 256 * 512 * 4);
  bf16_t* attn  = (bf16_t*)alloc((size_t)256 * 256 * 512 * 2);
  float*  gbuf  = (float*)alloc((size_t)8192 * 512 * 4);
  float*  tbuf  = (float*)alloc((size_t)8192 * 512 * 4);
  float*  cur   = (float*)alloc((size_t)8192 * 512 * 4);
  bf16_t* abf   = (bf16_t*)alloc((size_t)8192 * 512 * 2);
  bf16_t* ffh   = (bf16_t*)score;   // alias: score dead after softmax, reborn next layer
  const int BIG = 1 << 30;

  auto tr = [&](const float* in, bf16_t* out, int R, int C) {
    long long n = (long long)R * C;
    transpose_bf_k<<<dim3((unsigned)((n + 255) / 256)), dim3(256), 0, stream>>>(in, out, R, C);
  };

  // ---- weight prep (f32 -> bf16, [K,N] -> [N,K]) ----
  tr(stem_w1, w1T, 576, 512);
  tr(stem_w2, w2T, 512, 512);
  for (int l = 0; l < 4; ++l) {
    tr(Wqkv + (size_t)l * 512 * 1536, qkvT + (size_t)l * 1536 * 512, 512, 1536);
    tr(Wr   + (size_t)l * 512 * 512,  wrT  + (size_t)l * 512 * 512,  512, 512);
    tr(Wo   + (size_t)l * 512 * 512,  woT  + (size_t)l * 512 * 512,  512, 512);
    tr(W1   + (size_t)l * 512 * 2048, w1lT + (size_t)l * 2048 * 512, 512, 2048);
    tr(W2   + (size_t)l * 2048 * 512, w2lT + (size_t)l * 512 * 2048, 2048, 512);
  }

  // ---- stem ----
  build_stem_k<<<dim3((8192 * 576) / 256), dim3(256), 0, stream>>>(x, action, act_emb, X0);
  posemb_k<<<dim3(512), dim3(256), 0, stream>>>(positions, pe);

  run_gemm<EpiF32>(stream, 8192, 512, 576, 1, X0, 576, 0, BIG, w1T, 576, 0, BIG,
                   EpiF32::Params{gbuf, 512, 0, nullptr});
  ln_k<<<dim3(8192), dim3(256), 0, stream>>>(gbuf, nullptr, s_ln1_g, s_ln1_b, 1e-5f, 1, nullptr, abf);
  run_gemm<EpiF32>(stream, 8192, 512, 512, 1, abf, 512, 0, BIG, w2T, 512, 0, BIG,
                   EpiF32::Params{gbuf, 512, 0, nullptr});
  ln_k<<<dim3(8192), dim3(256), 0, stream>>>(gbuf, nullptr, s_ln2_g, s_ln2_b, 1e-5f, 0, tbuf, nullptr);
  ln_k<<<dim3(8192), dim3(256), 0, stream>>>(tbuf, nullptr, lnf_g, lnf_b, 1e-6f, 0, cur, nullptr);

  // ---- transformer layers ----
  for (int l = 0; l < 4; ++l) {
    build_cat_k<<<dim3((16384 * 512) / 256), dim3(256), 0, stream>>>(
        mems + (size_t)l * 256 * 32 * 512, cur, catbf);

    run_gemm<EpiQKV>(stream, 16384, 1536, 512, 1, catbf, 512, 0, BIG,
                     qkvT + (size_t)l * 1536 * 512, 512, 0, BIG,
                     EpiQKV::Params{qu, qv, kk, vT, u_bias, v_bias});

    run_gemm<EpiBF16>(stream, 512, 512, 512, 1, pe, 512, 0, BIG,
                      wrT + (size_t)l * 512 * 512, 512, 0, BIG,
                      EpiBF16::Params{rbf, 512});

    // AC: per (b,h): [256,64] x [512,64]^T -> score
    run_gemm<EpiF32>(stream, 256, 512, 64, 256,
                     qu, 64, 256 * 64, BIG, kk, 64, 512 * 64, BIG,
                     EpiF32::Params{score, 512, 256 * 512, nullptr});
    // BD: per (b,h): [256,64] x r_h[512,64]^T, rel-shift accumulated into score
    run_gemm<EpiBD>(stream, 256, 512, 64, 256,
                    qv, 64, 256 * 64, BIG, rbf, 512, 64, 8,
                    EpiBD::Params{score});

    softmax_k<<<dim3(256, 256), dim3(256), 0, stream>>>(score, attn, 0.125f);

    // attn @ V -> ao rows for Wo
    run_gemm<EpiAO>(stream, 256, 64, 512, 256,
                    attn, 512, 256 * 512, BIG, vT, 512, 64 * 512, BIG,
                    EpiAO::Params{abf});

    run_gemm<EpiF32>(stream, 8192, 512, 512, 1, abf, 512, 0, BIG,
                     woT + (size_t)l * 512 * 512, 512, 0, BIG,
                     EpiF32::Params{gbuf, 512, 0, nullptr});

    // h1 = LN(cur + attn_out)
    ln_k<<<dim3(8192), dim3(256), 0, stream>>>(gbuf, cur, ln1_g + l * 512, ln1_b + l * 512,
                                               1e-5f, 0, tbuf, abf);
    // FFN
    run_gemm<EpiReluBias>(stream, 8192, 2048, 512, 1, abf, 512, 0, BIG,
                          w1lT + (size_t)l * 2048 * 512, 512, 0, BIG,
                          EpiReluBias::Params{ffh, 2048, b1 + l * 2048});
    run_gemm<EpiF32>(stream, 8192, 512, 2048, 1, ffh, 2048, 0, BIG,
                     w2lT + (size_t)l * 512 * 2048, 2048, 0, BIG,
                     EpiF32::Params{gbuf, 512, 0, b2 + l * 512});
    // cur = LN(h1 + ff)
    ln_k<<<dim3(8192), dim3(256), 0, stream>>>(gbuf, tbuf, ln2_g + l * 512, ln2_b + l * 512,
                                               1e-5f, 0, cur, nullptr);
  }

  writeout_k<<<dim3((8192 * 512) / 256), dim3(256), 0, stream>>>(cur, (float*)d_out);
}
